// SeqNetSVHN_45930380263775
// MI455X (gfx1250) — matrix operand order repaired and checked by reading, not compiled, not run on hardware
//
#include <hip/hip_runtime.h>
#include <utility>

// ---------------------------------------------------------------------------
// SNN (norse LIF) forward:
//   cur = x @ w0^T + b0   (T*B=51200, F=3072, H=200)   -> bf16 WMMA GEMM
//   sequential LIF scan over T=100                      -> register-resident
// ---------------------------------------------------------------------------

typedef __attribute__((ext_vector_type(16))) __bf16 v16bf;
typedef __attribute__((ext_vector_type(8)))  float  v8f;

union FragBF {
    unsigned int u[8];
    uint4        q[2];
    v16bf        v;
};

__device__ __forceinline__ unsigned short f2bf(float f) {
    unsigned int u = __float_as_uint(f);
    unsigned int r = u + 0x7FFFu + ((u >> 16) & 1u);   // round-to-nearest-even
    return (unsigned short)(r >> 16);
}

// Compile-time unroller: gives each iteration a constexpr index so
// sched_group_barrier arguments are frontend-constant.
template <typename F, int... I>
__device__ __forceinline__ void ct_unroll_impl(F&& f, std::integer_sequence<int, I...>) {
    (f(std::integral_constant<int, I>{}), ...);
}
template <int N, typename F>
__device__ __forceinline__ void ct_unroll(F&& f) {
    ct_unroll_impl(static_cast<F&&>(f), std::make_integer_sequence<int, N>{});
}

// Scheduling pipeline hints (no-op if builtin unavailable).
#if defined(__has_builtin)
#if __has_builtin(__builtin_amdgcn_sched_group_barrier)
#define SGB(mask, cnt, id) __builtin_amdgcn_sched_group_barrier((mask), (cnt), (id))
#else
#define SGB(mask, cnt, id)
#endif
#else
#define SGB(mask, cnt, id)
#endif
#define SGB_DSREAD  0x100
#define SGB_WMMA    0x008
#define SGB_VMREAD  0x020

#define M_TOT   51200          // T*B
#define K_TOT   3072           // F
#define N_PAD   208            // H=200 padded to 13*16
#define BM      128            // CTA M tile (8 waves x 16 rows)
#define BK      64             // K stage depth
#define NT      13             // N tiles of 16

// ---------------------------------------------------------------------------
// Kernel 1: convert w0 [200,3072] f32 -> [208,3072] bf16 (zero-pad rows)
// ---------------------------------------------------------------------------
__global__ __launch_bounds__(256) void conv_w0_kernel(
        const float* __restrict__ w0, unsigned short* __restrict__ w0b) {
    int idx = blockIdx.x * 256 + threadIdx.x;       // 208*3072 = 638976 total
    int row = idx / K_TOT;
    int k   = idx - row * K_TOT;
    float v = (row < 200) ? w0[row * K_TOT + k] : 0.0f;
    w0b[idx] = f2bf(v);
}

// ---------------------------------------------------------------------------
// Kernel 2: GEMM  cur[m,n] = sum_k x[m,k]*w0[n,k] + b0[n]
//   x: fp32 [51200,3072], w0b: bf16 [208,3072], cur: fp32 [51200,208]
//   Pipelining: global->reg prefetch across K stages (single LDS buffer),
//   3-deep rotating B-fragment buffer + sched_group_barrier pipeline keeping
//   DS reads exactly 2 WMMAs ahead; fragments load as 2x ds_load_b128.
// ---------------------------------------------------------------------------
__global__ __launch_bounds__(256) void gemm_wmma_kernel(
        const float* __restrict__ x,
        const unsigned short* __restrict__ w0b,
        const float* __restrict__ b0,
        float* __restrict__ cur) {
    __shared__ __align__(16) unsigned short Alds[BM * BK];     // 16 KB
    __shared__ __align__(16) unsigned short Blds[N_PAD * BK];  // 26 KB

    const int tid  = threadIdx.x;
    const int wave = tid >> 5;
    const int lane = tid & 31;
    const int m0   = blockIdx.x * BM;

    v8f acc[NT];
#pragma unroll
    for (int i = 0; i < NT; ++i)
#pragma unroll
        for (int j = 0; j < 8; ++j) acc[i][j] = 0.0f;

    const unsigned int* Au = (const unsigned int*)Alds;
    const unsigned int* Bu = (const unsigned int*)Blds;

    // Fragment gathers (documented CDNA5 16-bit WMMA VGPR layouts).
    // Both halves are 16B-aligned contiguous runs -> exactly 2x ds_load_b128.
    auto loadA = [&](FragBF& fr, int kk) {
        const int m    = wave * 16 + (lane & 15);
        const int base = m * (BK / 2) + ((kk + (lane >> 4) * 8) >> 1);
        fr.q[0] = *(const uint4*)&Au[base];       // K pairs {0..3}   (+kb)
        fr.q[1] = *(const uint4*)&Au[base + 8];   // K pairs {8..11}  (+kb)
    };
    auto loadB = [&](FragBF& fr, int kk, int nt) {
        const int n    = nt * 16 + (lane & 15);
        const int base = n * (BK / 2) + ((kk + (lane >> 4) * 8) >> 1);
        fr.q[0] = *(const uint4*)&Bu[base];
        fr.q[1] = *(const uint4*)&Bu[base + 8];
    };

    // Global-stage registers (next K-slab held in VGPRs during compute).
    float4 aReg[8];
    uint2  bReg[13];

    auto globLoad = [&](int kt) {
#pragma unroll
        for (int i = 0; i < 8; ++i) {
            int f4  = tid + i * 256;
            int row = f4 >> 4;               // 16 float4 per 64-wide row
            int kc  = (f4 & 15) << 2;
            aReg[i] = *(const float4*)(x + (long)(m0 + row) * K_TOT + kt + kc);
        }
#pragma unroll
        for (int i = 0; i < 13; ++i) {
            int e   = tid + i * 256;
            int row = e >> 4;                // 16 x (4 bf16) per 64-wide row
            int kc  = (e & 15) << 2;
            bReg[i] = *(const uint2*)(w0b + (long)row * K_TOT + kt + kc);
        }
    };
    auto stageStore = [&]() {
#pragma unroll
        for (int i = 0; i < 8; ++i) {
            int f4  = tid + i * 256;
            int row = f4 >> 4;
            int kc  = (f4 & 15) << 2;
            uint2 p;
            p.x = (unsigned int)f2bf(aReg[i].x) | ((unsigned int)f2bf(aReg[i].y) << 16);
            p.y = (unsigned int)f2bf(aReg[i].z) | ((unsigned int)f2bf(aReg[i].w) << 16);
            *(uint2*)&Alds[row * BK + kc] = p;
        }
#pragma unroll
        for (int i = 0; i < 13; ++i) {
            int e   = tid + i * 256;
            int row = e >> 4;
            int kc  = (e & 15) << 2;
            *(uint2*)&Blds[row * BK + kc] = bReg[i];
        }
    };

    globLoad(0);                       // prologue: stage 0 -> regs

    for (int kt = 0; kt < K_TOT; kt += BK) {
        __syncthreads();               // previous compute done reading LDS
        stageStore();                  // regs -> LDS
        __syncthreads();
        if (kt + BK < K_TOT) globLoad(kt + BK);   // HBM latency hidden by WMMAs

        // ---- compute: 26 WMMAs, B fragments prefetched 2 steps ahead
        FragBF afr[2], bfr[3];
        loadA(afr[0], 0);
        loadB(bfr[0], 0, 0);
        loadB(bfr[1], 0, 1);

        ct_unroll<26>([&](auto stepc) {
            constexpr int step = decltype(stepc)::value;
            constexpr int nt   = (step < 13) ? step : step - 13;
            constexpr int ns   = step + 2;          // distance-2 prefetch
            if constexpr (ns < 26) {
                constexpr int nkk = (ns < 13) ? 0 : 32;
                constexpr int nnt = (ns < 13) ? ns : ns - 13;
                loadB(bfr[ns % 3], nkk, nnt);
                if constexpr (step == 11) loadA(afr[1], 32);  // second A frag
            }
            acc[nt] = __builtin_amdgcn_wmma_f32_16x16x32_bf16(
                false, afr[(step < 13) ? 0 : 1].v,
                false, bfr[step % 3].v,
                (short)0, acc[nt], false, false);

            // Scheduling pipeline: keep DS reads 2 WMMAs ahead; spread the 21
            // global prefetch loads across the WMMA chain.
            if constexpr (step == 0) { SGB(SGB_DSREAD, 6, 0); }  // A0+B0+B1
            constexpr int nds = (step < 24) ? ((step == 11) ? 4 : 2) : 0;
            if constexpr (nds > 0) { SGB(SGB_DSREAD, nds, 0); }
            SGB(SGB_WMMA, 1, 0);
            if constexpr (step < 21) { SGB(SGB_VMREAD, 1, 0); }
        });
    }

    // ---- store C (+bias). C layout: VGPR r -> M = r + 8*(lane>=16), N = lane&15
    const long outBase = (long)m0 * N_PAD;
    const int  mrow    = wave * 16 + ((lane >> 4) * 8);
#pragma unroll
    for (int nt = 0; nt < NT; ++nt) {
        int col    = nt * 16 + (lane & 15);
        float bias = (col < 200) ? b0[col] : 0.0f;
#pragma unroll
        for (int r = 0; r < 8; ++r) {
            cur[outBase + (long)(mrow + r) * N_PAD + col] = acc[nt][r] + bias;
        }
    }
}

// ---------------------------------------------------------------------------
// Kernel 3: sequential LIF scan.  Block = 8 batch rows (1 wave each).
//   Per lane: 7 hidden states (h = lane + 32k).  w_out staged in LDS.
// ---------------------------------------------------------------------------
__global__ __launch_bounds__(256) void lif_scan_kernel(
        const float* __restrict__ cur,
        const float* __restrict__ w_out,
        float* __restrict__ out) {
    __shared__ float wlds[2048];             // [10][200] padded, zero tail
    for (int i = threadIdx.x; i < 2048; i += 256)
        wlds[i] = (i < 2000) ? w_out[i] : 0.0f;
    __syncthreads();

    const int r    = threadIdx.x >> 5;
    const int lane = threadIdx.x & 31;
    const int b    = blockIdx.x * 8 + r;

    float vst[7], ist[7];
#pragma unroll
    for (int k = 0; k < 7; ++k) { vst[k] = 0.0f; ist[k] = 0.0f; }
    float vo[10], io[10];
#pragma unroll
    for (int o = 0; o < 10; ++o) { vo[o] = 0.0f; io[o] = 0.0f; }

    for (int t = 0; t < 100; ++t) {
        const float* crow = cur + ((long)t * 512 + b) * N_PAD;
        float p[10];
#pragma unroll
        for (int o = 0; o < 10; ++o) p[o] = 0.0f;

#pragma unroll
        for (int k = 0; k < 7; ++k) {
            const int h = lane + 32 * k;
            const float c  = (h < 200) ? crow[h] : 0.0f;
            const float vd = 0.9f * vst[k] + 0.1f * ist[k];   // v decay + input
            const float id = 0.8f * ist[k];                   // i decay
            const float z  = (vd > 1.0f) ? 1.0f : 0.0f;       // spike
            vst[k] = (1.0f - z) * vd;                         // reset to 0
            ist[k] = id + c;                                  // inject current
#pragma unroll
            for (int o = 0; o < 10; ++o) p[o] += z * wlds[o * 200 + h];
        }
        // wave32 butterfly reduce each output channel
#pragma unroll
        for (int o = 0; o < 10; ++o) {
            float s = p[o];
            s += __shfl_xor(s, 16, 32);
            s += __shfl_xor(s,  8, 32);
            s += __shfl_xor(s,  4, 32);
            s += __shfl_xor(s,  2, 32);
            s += __shfl_xor(s,  1, 32);
            p[o] = s;
        }
        if (lane == 0) {
            float* orow = out + ((long)t * 512 + b) * 10;
#pragma unroll
            for (int o = 0; o < 10; ++o) {
                io[o] *= 0.8f;                       // io_dec
                vo[o]  = 0.9f * vo[o] + 0.1f * io[o];// von
                io[o] += p[o];                       // ion = io_dec + z@w_out^T
                orow[o] = vo[o];
            }
        }
    }
}

// ---------------------------------------------------------------------------
extern "C" void kernel_launch(void* const* d_in, const int* in_sizes, int n_in,
                              void* d_out, int out_size, void* d_ws, size_t ws_size,
                              hipStream_t stream) {
    const float* x     = (const float*)d_in[0];   // [100,512,3072]
    const float* w0    = (const float*)d_in[1];   // [200,3072]
    const float* b0    = (const float*)d_in[2];   // [200]
    const float* w_out = (const float*)d_in[3];   // [10,200]
    float* out = (float*)d_out;                   // [100,512,10]

    unsigned short* w0b = (unsigned short*)d_ws;                       // 208*3072*2 B
    float* cur = (float*)((char*)d_ws + (size_t)N_PAD * K_TOT * 2);    // 51200*208*4 B

    conv_w0_kernel<<<(N_PAD * K_TOT) / 256, 256, 0, stream>>>(w0, w0b);
    gemm_wmma_kernel<<<M_TOT / BM, 256, 0, stream>>>(x, w0b, b0, cur);
    lif_scan_kernel<<<512 / 8, 256, 0, stream>>>(cur, w_out, out);
}
